// AttentionModel_39848706573312
// MI455X (gfx1250) — matrix operand order repaired and checked by reading, not compiled, not run on hardware
//
#include <hip/hip_runtime.h>
#include <hip/hip_bf16.h>

typedef __attribute__((ext_vector_type(16))) __bf16 v16bf;
typedef __attribute__((ext_vector_type(8)))  __bf16 v8bf;
typedef __attribute__((ext_vector_type(4)))  __bf16 v4bf;
typedef __attribute__((ext_vector_type(8)))  float  v8f;

union V16 { v16bf v; v8bf h[2]; };

#define S_LEN 2048
#define NHEAD 16
#define NBATCH 2
#define DHEAD 64
#define BHCNT (NBATCH * NHEAD)

// ---------------------------------------------------------------------------
// Kernel 1: QKV projection (x @ W.T + b) -> bf16 workspace.
//   Qw: [bh][s][d] row-major, Kw: [bh][s][d], Vw: [bh][d][s] transposed.
// WMMA results staged COLUMN-MAJOR in LDS (os[e*64+s]) so each lane's 8
// accumulator values store as one ds_store_b128; then cooperative b128 stores.
// ---------------------------------------------------------------------------
__global__ __launch_bounds__(256)
void qkv_proj_kernel(const float* __restrict__ q_in, const float* __restrict__ k_in,
                     const float* __restrict__ v_in,
                     const float* __restrict__ Wq, const float* __restrict__ bq,
                     const float* __restrict__ Wk, const float* __restrict__ bk,
                     const float* __restrict__ Wv, const float* __restrict__ bv,
                     __bf16* __restrict__ Qw, __bf16* __restrict__ Kw,
                     __bf16* __restrict__ Vw)
{
    __shared__ alignas(32) __bf16 xs[3][64 * 64];    // X tiles, row-major bf16
    __shared__ alignas(32) __bf16 wsm[3][64 * 64];   // W row-major: wsm[e*64+d]=W[e][d]
    __shared__ alignas(32) __bf16 os[64 * 64];       // staging, COLUMN-major: os[e*64+s]
    __shared__ float bs[3][64];

    const int tid  = threadIdx.x;
    const int tile = blockIdx.x & 31;        // S/64 = 32 tiles
    const int bh   = blockIdx.x >> 5;        // 0..31
    const int b    = bh >> 4;
    const int h    = bh & 15;
    const int s0   = tile * 64;

    const float* xin[3] = {q_in, k_in, v_in};
    const float* win[3] = {Wq, Wk, Wv};
    const float* bin[3] = {bq, bk, bv};

    // ---- load phase: b128 global loads, bf16 convert into LDS ----
    #pragma unroll
    for (int p = 0; p < 3; ++p) {
        for (int idx = tid; idx < 64 * 16; idx += 256) {      // 1024 float4 chunks
            const int r  = idx >> 4;
            const int c4 = (idx & 15) << 2;
            const float4 f = *(const float4*)(xin[p] +
                ((((size_t)(b * S_LEN + s0 + r)) * NHEAD + h) << 6) + c4);
            v4bf t4 = { (__bf16)f.x, (__bf16)f.y, (__bf16)f.z, (__bf16)f.w };
            *(v4bf*)&xs[p][r * 64 + c4] = t4;
            // weights: read W[e][d0..d0+3] contiguously, store row-major to LDS
            const int e  = idx >> 4;
            const int d4 = (idx & 15) << 2;
            const float4 w4 = *(const float4*)(win[p] + (e << 6) + d4);
            wsm[p][e * 64 + d4 + 0] = (__bf16)w4.x;
            wsm[p][e * 64 + d4 + 1] = (__bf16)w4.y;
            wsm[p][e * 64 + d4 + 2] = (__bf16)w4.z;
            wsm[p][e * 64 + d4 + 3] = (__bf16)w4.w;
        }
        if (tid < 64) bs[p][tid] = bin[p][tid];
    }
    __syncthreads();

    const int wave  = tid >> 5;
    const int lane  = tid & 31;
    const int lhalf = lane >> 4;
    const int l16   = lane & 15;

    #pragma unroll
    for (int p = 0; p < 3; ++p) {
        // ---- compute: 16 output 16x16 tiles, 2 per wave ----
        #pragma unroll
        for (int ti = 0; ti < 2; ++ti) {
            const int t  = wave + ti * 8;
            const int m0 = (t >> 2) * 16;
            const int e0 = (t & 3) * 16;
            v8f acc = {};
            #pragma unroll
            for (int kk = 0; kk < 2; ++kk) {                  // D=64 -> 2 chunks of 32
                V16 a, bm;
                const int base = 32 * kk + 8 * lhalf;
                a.h[0] = *(const v8bf*)&xs[p][(m0 + l16) * 64 + base];
                a.h[1] = *(const v8bf*)&xs[p][(m0 + l16) * 64 + base + 16];
                bm.h[0] = *(const v8bf*)&wsm[p][(e0 + l16) * 64 + base];
                bm.h[1] = *(const v8bf*)&wsm[p][(e0 + l16) * 64 + base + 16];
                acc = __builtin_amdgcn_wmma_f32_16x16x32_bf16(
                        false, a.v, false, bm.v, (short)0, acc, false, false);
            }
            // lane owns column e0+l16, rows m0+8*lhalf..+8 -> contiguous in os
            const float bias = bs[p][e0 + l16];
            v8bf ov;
            #pragma unroll
            for (int r = 0; r < 8; ++r) ov[r] = (__bf16)(acc[r] + bias);
            *(v8bf*)&os[(e0 + l16) * 64 + m0 + 8 * lhalf] = ov;   // one b128 store
        }
        __syncthreads();

        // ---- cooperative b128 stores from staging LDS (column-major os) ----
        if (p == 2) {
            // Vw transposed [bh][d][s]: os is already (e,s) -> fully vectorized
            #pragma unroll
            for (int c = 0; c < 2; ++c) {
                const int chunk = tid + c * 256;              // 512 chunks of 8
                const int e  = chunk >> 3;
                const int s8 = (chunk & 7) << 3;
                *(v8bf*)(Vw + ((size_t)bh * 64 + e) * S_LEN + s0 + s8) =
                    *(const v8bf*)&os[e * 64 + s8];
            }
        } else {
            // Q/K row-major [bh][s][d]: gather 8 strided bf16 from os, b128 store
            __bf16* const dst = (p == 0) ? Qw : Kw;
            #pragma unroll
            for (int c = 0; c < 2; ++c) {
                const int chunk = tid + c * 256;
                const int row = chunk >> 3;
                const int c8  = (chunk & 7) << 3;
                v8bf pk;
                #pragma unroll
                for (int i = 0; i < 8; ++i) pk[i] = os[(c8 + i) * 64 + row];
                *(v8bf*)(dst + ((size_t)bh * S_LEN + s0 + row) * 64 + c8) = pk;
            }
        }
        __syncthreads();   // os reused by next projection
    }
}

// ---------------------------------------------------------------------------
// Kernel 2: flash attention. Block = 8 waves, each wave owns 16 query rows
// (128 rows/block). Online softmax; dropout applied to unnormalized P;
// C->A transpose of P via per-wave LDS slice.
// ---------------------------------------------------------------------------
__global__ __launch_bounds__(256)
void flash_attn_kernel(const __bf16* __restrict__ Qw, const __bf16* __restrict__ Kw,
                       const __bf16* __restrict__ Vw, const float* __restrict__ mask,
                       const float* __restrict__ inv_scale_p, float* __restrict__ out)
{
    __shared__ alignas(32) __bf16 p_lds[8][16 * 64];   // per-wave P tile (bf16)

    const int tid   = threadIdx.x;
    const int wave  = tid >> 5;
    const int lane  = tid & 31;
    const int lhalf = lane >> 4;
    const int l16   = lane & 15;

    const int qtile = blockIdx.x & 15;       // S/128 = 16 query tiles
    const int bh    = blockIdx.x >> 4;       // 0..31
    const int b     = bh >> 4;
    const int qr0   = qtile * 128 + wave * 16;

    const float scale = 1.0f / inv_scale_p[0];   // reference divides by inv_scale

    // Q A-fragments, kept in registers for the whole pass
    V16 aq[2];
    const __bf16* qbase = Qw + ((size_t)bh * S_LEN + qr0 + l16) * 64;
    #pragma unroll
    for (int kk = 0; kk < 2; ++kk) {
        const int base = 32 * kk + 8 * lhalf;
        aq[kk].h[0] = *(const v8bf*)(qbase + base);
        aq[kk].h[1] = *(const v8bf*)(qbase + base + 16);
    }

    v8f o[4] = {};                 // 16 x 64 output accumulator (4 N-tiles)
    float m8[8], l8[8];            // per-row running max / sumexp (rows r+8*lhalf)
    #pragma unroll
    for (int r = 0; r < 8; ++r) { m8[r] = -1e30f; l8[r] = 0.0f; }

    for (int t0 = 0; t0 < S_LEN; t0 += 64) {
        // prefetch next K/V tiles into cache (global_prefetch path)
        if (t0 + 64 < S_LEN) {
            __builtin_prefetch(Kw + ((size_t)bh * 64 + l16) * S_LEN + t0 + 64, 0, 1);
            __builtin_prefetch(Vw + ((size_t)bh * S_LEN + t0 + 64 + lane) * 64, 0, 1);
        }

        // ---- scores S = Q K^T : 4 tiles of 16x16, contraction D=64 ----
        v8f sc[4];
        #pragma unroll
        for (int j = 0; j < 4; ++j) {
            v8f c = {};
            #pragma unroll
            for (int kk = 0; kk < 2; ++kk) {
                V16 bk;                               // lane = token, elems = 16 d
                const __bf16* kp = Kw + ((size_t)bh * S_LEN + t0 + 16 * j + l16) * 64 + 32 * kk + 8 * lhalf;
                bk.h[0] = *(const v8bf*)(kp);
                bk.h[1] = *(const v8bf*)(kp + 16);
                c = __builtin_amdgcn_wmma_f32_16x16x32_bf16(
                        false, aq[kk].v, false, bk.v, (short)0, c, false, false);
            }
            sc[j] = c;
        }

        // ---- scale + mask + row max ----
        float mt[8];
        #pragma unroll
        for (int r = 0; r < 8; ++r) mt[r] = -1e30f;
        #pragma unroll
        for (int j = 0; j < 4; ++j) {
            const int tg = t0 + 16 * j + l16;
            #pragma unroll
            for (int r = 0; r < 8; ++r) {
                const int row = qr0 + r + 8 * lhalf;
                const float v = sc[j][r] * scale +
                                mask[(size_t)b * S_LEN * S_LEN + (size_t)row * S_LEN + tg];
                sc[j][r] = v;
                mt[r] = fmaxf(mt[r], v);
            }
        }
        #pragma unroll
        for (int r = 0; r < 8; ++r) {                 // reduce over 16 lanes (N dim)
            float v = mt[r];
            v = fmaxf(v, __shfl_xor(v, 1));
            v = fmaxf(v, __shfl_xor(v, 2));
            v = fmaxf(v, __shfl_xor(v, 4));
            v = fmaxf(v, __shfl_xor(v, 8));
            mt[r] = fmaxf(v, m8[r]);
        }

        // ---- P = exp(S - m), row sums (pre-dropout), rescale accumulators ----
        float rs[8];
        #pragma unroll
        for (int r = 0; r < 8; ++r) rs[r] = 0.0f;
        #pragma unroll
        for (int j = 0; j < 4; ++j) {
            #pragma unroll
            for (int r = 0; r < 8; ++r) {
                const float p = __expf(sc[j][r] - mt[r]);
                sc[j][r] = p;
                rs[r] += p;
            }
        }
        #pragma unroll
        for (int r = 0; r < 8; ++r) {
            float v = rs[r];
            v += __shfl_xor(v, 1);
            v += __shfl_xor(v, 2);
            v += __shfl_xor(v, 4);
            v += __shfl_xor(v, 8);
            const float alpha = __expf(m8[r] - mt[r]);
            l8[r] = l8[r] * alpha + v;
            m8[r] = mt[r];
            #pragma unroll
            for (int j = 0; j < 4; ++j) o[j][r] *= alpha;
        }

        // ---- dropout on unnormalized P, write to LDS (C layout -> row major) ----
        #pragma unroll
        for (int j = 0; j < 4; ++j) {
            const int tg = t0 + 16 * j + l16;
            #pragma unroll
            for (int r = 0; r < 8; ++r) {
                const int row = qr0 + r + 8 * lhalf;
                unsigned idx = (unsigned)(((unsigned)(bh * S_LEN + row)) * (unsigned)S_LEN + tg);
                unsigned hsh = idx * 2654435761u;
                hsh ^= hsh >> 16; hsh *= 0x85ebca6bu; hsh ^= hsh >> 13;
                float p = sc[j][r];
                p = (hsh < 429496730u) ? 0.0f : p * (1.0f / 0.9f);   // p_drop = 0.1
                p_lds[wave][(r + 8 * lhalf) * 64 + 16 * j + l16] = (__bf16)p;
            }
        }
        asm volatile("s_wait_dscnt 0" ::: "memory");

        // ---- O += P V : P as A-fragment from LDS, V as B-fragment from global ----
        V16 ap[2];
        #pragma unroll
        for (int kk = 0; kk < 2; ++kk) {
            const int base = 32 * kk + 8 * lhalf;
            ap[kk].h[0] = *(const v8bf*)&p_lds[wave][l16 * 64 + base];
            ap[kk].h[1] = *(const v8bf*)&p_lds[wave][l16 * 64 + base + 16];
        }
        #pragma unroll
        for (int j = 0; j < 4; ++j) {
            #pragma unroll
            for (int kk = 0; kk < 2; ++kk) {
                V16 bvf;                              // lane = d column of V
                const __bf16* vp = Vw + ((size_t)bh * 64 + 16 * j + l16) * S_LEN + t0 + 32 * kk + 8 * lhalf;
                bvf.h[0] = *(const v8bf*)(vp);
                bvf.h[1] = *(const v8bf*)(vp + 16);
                o[j] = __builtin_amdgcn_wmma_f32_16x16x32_bf16(
                         false, ap[kk].v, false, bvf.v, (short)0, o[j], false, false);
            }
        }
    }

    // ---- epilogue: divide by softmax denominator, store fp32 [B,H,S,D] ----
    #pragma unroll
    for (int r = 0; r < 8; ++r) {
        const float inv_l = 1.0f / l8[r];
        const int row = qr0 + r + 8 * lhalf;
        #pragma unroll
        for (int j = 0; j < 4; ++j) {
            out[((size_t)bh * S_LEN + row) * 64 + 16 * j + l16] = o[j][r] * inv_l;
        }
    }
}

// ---------------------------------------------------------------------------
extern "C" void kernel_launch(void* const* d_in, const int* in_sizes, int n_in,
                              void* d_out, int out_size, void* d_ws, size_t ws_size,
                              hipStream_t stream) {
    const float* q_in  = (const float*)d_in[0];
    const float* k_in  = (const float*)d_in[1];
    const float* v_in  = (const float*)d_in[2];
    const float* amask = (const float*)d_in[3];
    const float* iscal = (const float*)d_in[4];
    const float* Wq    = (const float*)d_in[5];
    const float* bq    = (const float*)d_in[6];
    const float* Wk    = (const float*)d_in[7];
    const float* bk    = (const float*)d_in[8];
    const float* Wv    = (const float*)d_in[9];
    const float* bv    = (const float*)d_in[10];
    float* out = (float*)d_out;

    const size_t per = (size_t)BHCNT * S_LEN * DHEAD;   // bf16 elements per tensor
    __bf16* Qw = (__bf16*)d_ws;
    __bf16* Kw = Qw + per;
    __bf16* Vw = Kw + per;

    // 1 block per (b,h,64-token tile): 32 * 32 = 1024 blocks
    qkv_proj_kernel<<<dim3(BHCNT * (S_LEN / 64)), dim3(256), 0, stream>>>(
        q_in, k_in, v_in, Wq, bq, Wk, bk, Wv, bv, Qw, Kw, Vw);

    // 1 block per (b,h,128-row query tile): 32 * 16 = 512 blocks
    flash_attn_kernel<<<dim3(BHCNT * (S_LEN / 128)), dim3(256), 0, stream>>>(
        Qw, Kw, Vw, amask, iscal, out);
}
